// graph_ChebNet_51573967290506
// MI455X (gfx1250) — matrix operand order repaired and checked by reading, not compiled, not run on hardware
//
#include <hip/hip_runtime.h>

typedef __attribute__((ext_vector_type(16))) _Float16 v16h;
typedef __attribute__((ext_vector_type(8)))  _Float16 v8h;
typedef __attribute__((ext_vector_type(8)))  float    v8f;
typedef __attribute__((ext_vector_type(4)))  float    v4f;

#define N_NODES 50000
#define N_EDGES 320000
#define IN_DIM  48
#define HDIM    256

// ---------------- setup kernels ----------------

__global__ void k_zero_f32(float* __restrict__ p, int n) {
    int i = blockIdx.x * blockDim.x + threadIdx.x;
    if (i < n) p[i] = 0.0f;
}

__global__ void k_zero_i32(int* __restrict__ p, int n) {
    int i = blockIdx.x * blockDim.x + threadIdx.x;
    if (i < n) p[i] = 0;
}

// out-degree (float, for dinv) over src; in-degree histogram (int, for CSR) over dst
__global__ void k_degcnt(const long long* __restrict__ src, const long long* __restrict__ dst,
                         float* __restrict__ deg_src, int* __restrict__ cnt_dst) {
    int e = blockIdx.x * blockDim.x + threadIdx.x;
    if (e < N_EDGES) {
        atomicAdd(&deg_src[(int)src[e]], 1.0f);
        atomicAdd(&cnt_dst[(int)dst[e]], 1);
    }
}

__global__ void k_dinv(float* __restrict__ deg) {
    int i = blockIdx.x * blockDim.x + threadIdx.x;
    if (i < N_NODES) {
        float d = deg[i];
        deg[i] = (d > 0.0f) ? rsqrtf(fmaxf(d, 1.0f)) : 0.0f;
    }
}

// single-block exclusive scan of cnt_dst -> rowptr[N_NODES+1], also seeds cursor
__global__ __launch_bounds__(1024)
void k_scan(const int* __restrict__ cnt, int* __restrict__ rowptr, int* __restrict__ cursor) {
    __shared__ int buf[1024];
    __shared__ int carry;
    int tid = threadIdx.x;
    if (tid == 0) carry = 0;
    __syncthreads();
    for (int base = 0; base < N_NODES; base += 1024) {
        int i = base + tid;
        int v = (i < N_NODES) ? cnt[i] : 0;
        buf[tid] = v;
        __syncthreads();
        #pragma unroll
        for (int off = 1; off < 1024; off <<= 1) {
            int t = (tid >= off) ? buf[tid - off] : 0;
            __syncthreads();
            buf[tid] += t;
            __syncthreads();
        }
        int excl = carry + buf[tid] - v;
        if (i < N_NODES) { rowptr[i] = excl; cursor[i] = excl; }
        __syncthreads();
        if (tid == 1023) carry += buf[1023];
        __syncthreads();
    }
    if (tid == 0) rowptr[N_NODES] = carry;
}

// bucket-sort edges by dst: store src index and edge norm in CSR order
__global__ void k_fill(const long long* __restrict__ src, const long long* __restrict__ dst,
                       const float* __restrict__ dinv, int* __restrict__ cursor,
                       int* __restrict__ esrc, float* __restrict__ enorm) {
    int e = blockIdx.x * blockDim.x + threadIdx.x;
    if (e >= N_EDGES) return;
    int s = (int)src[e];
    int d = (int)dst[e];
    int pos = atomicAdd(&cursor[d], 1);
    esrc[pos]  = s;
    enorm[pos] = -dinv[s] * dinv[d];
}

// ---------------- per-node gather-reduce (no atomics) ----------------
// tx[d] = sum over incoming edges of norm * h[src]; one wave per node.
// Single edge-loop pass: each lane owns up to two float4 column chunks
// (cols lane*4 and (lane+32)*4). Edge src/weight are made wave-uniform via
// readfirstlane so row base addresses live in SGPRs (saddr-form b128 loads).
__global__ __launch_bounds__(256)
void k_aggregate(const float* __restrict__ h, const int* __restrict__ rowptr,
                 const int* __restrict__ esrc, const float* __restrict__ enorm,
                 float* __restrict__ tx, int K) {
    int lane = threadIdx.x & 31;
    int wave = threadIdx.x >> 5;
    int node = blockIdx.x * 8 + wave;
    if (node >= N_NODES) return;
    int beg = rowptr[node];
    int end = rowptr[node + 1];
    const int nch  = K >> 2;            // float4 chunks per row: 12 or 64
    const int c0   = lane * 4;
    const int c1   = (lane + 32) * 4;
    const bool has0 = lane < nch;
    const bool has1 = (lane + 32) < nch;

    v4f a0 = {0.0f, 0.0f, 0.0f, 0.0f};
    v4f a1 = {0.0f, 0.0f, 0.0f, 0.0f};

    for (int e = beg; e < end; ++e) {
        int   s = __builtin_amdgcn_readfirstlane(esrc[e]);
        float w = __uint_as_float(__builtin_amdgcn_readfirstlane(__float_as_uint(enorm[e])));
        const float* rowp = h + (size_t)s * K;
        if (has0) {
            v4f v = *(const v4f*)(rowp + c0);
            a0[0] = fmaf(w, v[0], a0[0]);
            a0[1] = fmaf(w, v[1], a0[1]);
            a0[2] = fmaf(w, v[2], a0[2]);
            a0[3] = fmaf(w, v[3], a0[3]);
        }
        if (has1) {
            v4f v = *(const v4f*)(rowp + c1);
            a1[0] = fmaf(w, v[0], a1[0]);
            a1[1] = fmaf(w, v[1], a1[1]);
            a1[2] = fmaf(w, v[2], a1[2]);
            a1[3] = fmaf(w, v[3], a1[3]);
        }
    }

    float* op = tx + (size_t)node * K;
    if (has0) *(v4f*)(op + c0) = a0;
    if (has1) *(v4f*)(op + c1) = a1;
}

// ---------------- weight conversion ----------------
// weights (2, K, HDIM) f32 -> Bt (HDIM, 2K) f16 (column-major of concat weight)
__global__ void k_wcat(const float* __restrict__ w, _Float16* __restrict__ bt, int K) {
    int K2 = 2 * K;
    int idx = blockIdx.x * blockDim.x + threadIdx.x;
    if (idx >= HDIM * K2) return;
    int n = idx / K2;
    int k = idx % K2;
    int part = (k >= K) ? 1 : 0;
    int kk = k - part * K;
    bt[idx] = (_Float16)w[((size_t)part * K + kk) * HDIM + n];
}

// ---------------- WMMA GEMM ----------------
// out(N_NODES x HDIM) = [A1 | A2](N_NODES x 2K) @ Wcat(2K x HDIM) + bias, opt ReLU.

__device__ inline v8h load8_f32_as_f16(const float* __restrict__ p) {
    v4f a = *(const v4f*)p;
    v4f b = *(const v4f*)(p + 4);
    v8h r;
    r[0] = (_Float16)a[0]; r[1] = (_Float16)a[1];
    r[2] = (_Float16)a[2]; r[3] = (_Float16)a[3];
    r[4] = (_Float16)b[0]; r[5] = (_Float16)b[1];
    r[6] = (_Float16)b[2]; r[7] = (_Float16)b[3];
    return r;
}

__global__ __launch_bounds__(256)
void k_gemm(const float* __restrict__ A1, const float* __restrict__ A2, int K,
            const _Float16* __restrict__ Bt, const float* __restrict__ bias,
            float* __restrict__ out, int relu) {
    const int K2   = 2 * K;
    const int lane = threadIdx.x & 31;
    const int wave = threadIdx.x >> 5;
    const int tile_m = blockIdx.x * 4 + (wave >> 1);
    if (tile_m * 16 >= N_NODES) return;            // wave-uniform: EXEC stays all-1s
    const int nbase = (wave & 1) * 8;              // 8 n-tiles of 16 columns each
    const int lm    = lane & 15;
    const int lhi   = lane >> 4;
    const int row   = tile_m * 16 + lm;

    v8f acc[8] = {};

    for (int ks = 0; ks < K2; ks += 32) {
        // A fragment: 16x32 f16; lane holds 16 contiguous halfs at c0 and c0+8
        int c0 = ks + lhi * 16;
        int c1 = c0 + 8;
        const float* p0 = (c0 < K) ? (A1 + (size_t)row * K + c0)
                                   : (A2 + (size_t)row * K + (c0 - K));
        const float* p1 = (c1 < K) ? (A1 + (size_t)row * K + c1)
                                   : (A2 + (size_t)row * K + (c1 - K));
        v8h alo = load8_f32_as_f16(p0);
        v8h ahi = load8_f32_as_f16(p1);
        v16h a;
        #pragma unroll
        for (int i = 0; i < 8; ++i) { a[i] = alo[i]; a[8 + i] = ahi[i]; }

        #pragma unroll
        for (int t = 0; t < 8; ++t) {
            // B fragment: 32x16 f16; lane = column n, 16 contiguous K values
            int n = (nbase + t) * 16 + lm;
            const _Float16* bp = Bt + (size_t)n * K2 + ks + lhi * 16;
            v8h b0 = *(const v8h*)bp;
            v8h b1 = *(const v8h*)(bp + 8);
            v16h b;
            #pragma unroll
            for (int i = 0; i < 8; ++i) { b[i] = b0[i]; b[8 + i] = b1[i]; }

            acc[t] = __builtin_amdgcn_wmma_f32_16x16x32_f16(
                false, a, false, b, (short)0, acc[t], false, false);
        }
    }

    // epilogue: C/D layout -> VGPR j holds M = j + lhi*8, N = lm
    const int mbase = tile_m * 16 + lhi * 8;
    #pragma unroll
    for (int t = 0; t < 8; ++t) {
        int n = (nbase + t) * 16 + lm;
        float bv = bias[n];
        #pragma unroll
        for (int j = 0; j < 8; ++j) {
            float v = acc[t][j] + bv;
            if (relu) v = fmaxf(v, 0.0f);
            out[(size_t)(mbase + j) * HDIM + n] = v;
        }
    }
}

// ---------------- host-side launch ----------------

static inline size_t align256(size_t x) { return (x + 255) & ~(size_t)255; }

extern "C" void kernel_launch(void* const* d_in, const int* in_sizes, int n_in,
                              void* d_out, int out_size, void* d_ws, size_t ws_size,
                              hipStream_t stream) {
    (void)in_sizes; (void)n_in; (void)out_size; (void)ws_size;

    const float*     x    = (const float*)d_in[0];
    const long long* edge = (const long long*)d_in[1];   // (2, N_EDGES) int64
    const float*     w1   = (const float*)d_in[2];
    const float*     b1   = (const float*)d_in[3];
    const float*     w2   = (const float*)d_in[4];
    const float*     b2   = (const float*)d_in[5];
    const float*     w3   = (const float*)d_in[6];
    const float*     b3   = (const float*)d_in[7];
    float* out = (float*)d_out;

    const long long* src = edge;
    const long long* dst = edge + N_EDGES;

    // workspace carve-up
    char* ws = (char*)d_ws;
    size_t off = 0;
    float* deg    = (float*)(ws + off); off += align256((size_t)N_NODES * 4);  // -> dinv
    int*   cnt    = (int*)  (ws + off); off += align256((size_t)N_NODES * 4);
    int*   rowptr = (int*)  (ws + off); off += align256(((size_t)N_NODES + 1) * 4);
    int*   cursor = (int*)  (ws + off); off += align256((size_t)N_NODES * 4);
    int*   esrc   = (int*)  (ws + off); off += align256((size_t)N_EDGES * 4);
    float* enorm  = (float*)(ws + off); off += align256((size_t)N_EDGES * 4);
    _Float16* bt1 = (_Float16*)(ws + off); off += align256((size_t)HDIM * 2 * IN_DIM * 2);
    _Float16* bt2 = (_Float16*)(ws + off); off += align256((size_t)HDIM * 2 * HDIM * 2);
    _Float16* bt3 = (_Float16*)(ws + off); off += align256((size_t)HDIM * 2 * HDIM * 2);
    float* tx = (float*)(ws + off); off += align256((size_t)N_NODES * HDIM * 4);
    float* h1 = (float*)(ws + off); off += align256((size_t)N_NODES * HDIM * 4);
    float* h2 = (float*)(ws + off); off += align256((size_t)N_NODES * HDIM * 4);

    const int T = 256;
    const int gemm_grid = (N_NODES / 16 + 3) / 4;    // 3125 row tiles, 4 per block
    const int agg_grid  = (N_NODES + 7) / 8;         // 8 waves (nodes) per block

    // degrees -> dinv; CSR build (count, scan, fill with norm)
    k_zero_f32<<<(N_NODES + T - 1) / T, T, 0, stream>>>(deg, N_NODES);
    k_zero_i32<<<(N_NODES + T - 1) / T, T, 0, stream>>>(cnt, N_NODES);
    k_degcnt<<<(N_EDGES + T - 1) / T, T, 0, stream>>>(src, dst, deg, cnt);
    k_dinv<<<(N_NODES + T - 1) / T, T, 0, stream>>>(deg);
    k_scan<<<1, 1024, 0, stream>>>(cnt, rowptr, cursor);
    k_fill<<<(N_EDGES + T - 1) / T, T, 0, stream>>>(src, dst, deg, cursor, esrc, enorm);

    // weight conversion to f16 Bt layout
    k_wcat<<<(HDIM * 2 * IN_DIM + T - 1) / T, T, 0, stream>>>(w1, bt1, IN_DIM);
    k_wcat<<<(HDIM * 2 * HDIM + T - 1) / T, T, 0, stream>>>(w2, bt2, HDIM);
    k_wcat<<<(HDIM * 2 * HDIM + T - 1) / T, T, 0, stream>>>(w3, bt3, HDIM);

    // layer 1: K = 48
    k_aggregate<<<agg_grid, T, 0, stream>>>(x, rowptr, esrc, enorm, tx, IN_DIM);
    k_gemm<<<gemm_grid, T, 0, stream>>>(x, tx, IN_DIM, bt1, b1, h1, 1);
    // layer 2: K = 256
    k_aggregate<<<agg_grid, T, 0, stream>>>(h1, rowptr, esrc, enorm, tx, HDIM);
    k_gemm<<<gemm_grid, T, 0, stream>>>(h1, tx, HDIM, bt2, b2, h2, 1);
    // layer 3: K = 256, no ReLU, write d_out
    k_aggregate<<<agg_grid, T, 0, stream>>>(h2, rowptr, esrc, enorm, tx, HDIM);
    k_gemm<<<gemm_grid, T, 0, stream>>>(h2, tx, HDIM, bt3, b3, out, 0);
}
